// PerceptionAgent_21818433864233
// MI455X (gfx1250) — matrix operand order repaired and checked by reading, not compiled, not run on hardware
//
#include <hip/hip_runtime.h>
#include <hip/hip_bf16.h>
#include <math.h>

typedef __bf16 bf16_t;
typedef __attribute__((ext_vector_type(16))) __bf16 v16bf;
typedef __attribute__((ext_vector_type(8)))  __bf16 v8bf;
typedef __attribute__((ext_vector_type(8)))  float  v8f;
typedef __attribute__((ext_vector_type(4)))  unsigned int v4u;
typedef __attribute__((ext_vector_type(4)))  int v4i;
typedef __attribute__((ext_vector_type(8)))  int v8i;

#define S_LEN   2048
#define BATCH   2
#define INPUT_D 512
#define HID_D   1024
#define OUT_D   256
#define HEADS_N 16
#define HEAD_D  64

// ---------------------------------------------------------------- conversions
__global__ void cvt_f32_bf16(const float* __restrict__ in, bf16_t* __restrict__ out, int n) {
  int i = blockIdx.x * blockDim.x + threadIdx.x;
  int stride = gridDim.x * blockDim.x;
  for (; i < n; i += stride) out[i] = (bf16_t)in[i];
}

// in: [K][N] f32, out: [N][K] bf16 (transposed so GEMM B-tiles are contiguous)
__global__ void cvt_f32_bf16_t(const float* __restrict__ in, bf16_t* __restrict__ out,
                               int K, int N) {
  int i = blockIdx.x * blockDim.x + threadIdx.x;
  int stride = gridDim.x * blockDim.x;
  int total = K * N;
  for (; i < total; i += stride) {
    int n = i / K, k = i - n * K;
    out[i] = (bf16_t)in[(size_t)k * N + n];
  }
}

// ---------------------------------------------------------------- TDM tile load
#define TM 128
#define TN 64
#define TK 32
#define LDA_P 40   // 32 + 8 bf16 pad (TDM: pad 4 DWORDs per 16 DWORDs)
#define LDB_P 40

#if defined(__has_builtin)
#if __has_builtin(__builtin_amdgcn_tensor_load_to_lds)
#define USE_TDM 1
#endif
#endif
#ifndef USE_TDM
#define USE_TDM 0
#endif

#if USE_TDM
// 2-D tile DMA: tile (tile1 rows x tile0 elems) of a (tensorD1 x tensorD0) bf16 tensor,
// row stride strideD0 elems, into LDS at lds_off with 16B pad per 64B row.
__device__ __forceinline__ void tdm_load_tile(const bf16_t* gsrc, unsigned lds_off,
                                              unsigned tensorD0, unsigned tensorD1,
                                              unsigned tile0, unsigned tile1,
                                              unsigned strideD0) {
  unsigned long long ga = (unsigned long long)(size_t)gsrc;
  v4u g0;
  g0[0] = 1u;                                               // count=1, user mode
  g0[1] = lds_off;                                          // LDS byte address
  g0[2] = (unsigned)ga;                                     // global_addr[31:0]
  g0[3] = (unsigned)((ga >> 32) & 0x1FFFFFFull) | (2u << 30); // addr[56:32] | type=2

  // data_size=1 (2B) | pad_enable | pad_interval=3 (16 DW) | pad_amount=3 (4 DW)
  unsigned flags = (1u << 16) | (1u << 20) | (3u << 22) | (3u << 25);
  unsigned long long q0 = (unsigned long long)flags
                        | ((unsigned long long)(tensorD0 & 0xFFFFu) << 48);
  unsigned long long q1 = (unsigned long long)(tensorD0 >> 16)
                        | ((unsigned long long)tensorD1 << 16)
                        | ((unsigned long long)tile0 << 48);
  unsigned long long q2 = (unsigned long long)tile1
                        | ((unsigned long long)strideD0 << 32);
  unsigned long long q3 = 0ull;  // stride0[47:32]=0, tensor_dim1_stride=0 (2-D)
  v8i g1;
  g1[0] = (int)q0; g1[1] = (int)(q0 >> 32);
  g1[2] = (int)q1; g1[3] = (int)(q1 >> 32);
  g1[4] = (int)q2; g1[5] = (int)(q2 >> 32);
  g1[6] = (int)q3; g1[7] = (int)(q3 >> 32);
  v4i gz = {0, 0, 0, 0};
#if __clang_major__ >= 23
  v8i gz8 = {0, 0, 0, 0, 0, 0, 0, 0};
  __builtin_amdgcn_tensor_load_to_lds(g0, g1, gz, gz, gz8, 0);
#else
  __builtin_amdgcn_tensor_load_to_lds(g0, g1, gz, gz, 0);
#endif
}
#endif

// ---------------------------------------------------------------- bf16 WMMA GEMM
// C(bf16)[M,N] = A(bf16)[M,K] @ Bt(bf16)[N,K]^T + bias(f32)[N]
// Block tile 128x64, K-step 32, 256 threads = 8 waves. Double-buffered LDS;
// TDM DMA staging when available (wave 0 issues, TENSORcnt pipelines tiles).
__global__ __launch_bounds__(256) void gemm_bf16_wmma(
    const bf16_t* __restrict__ A, const bf16_t* __restrict__ Bt,
    const float* __restrict__ bias, bf16_t* __restrict__ C,
    int M, int N, int K) {
  __shared__ __align__(16) bf16_t sA[2][TM * LDA_P];
  __shared__ __align__(16) bf16_t sB[2][TN * LDB_P];

  const int tid   = threadIdx.x;
  const int lane  = tid & 31;
  const int wave  = tid >> 5;
  const int mBase = blockIdx.y * TM;
  const int nBase = blockIdx.x * TN;

  v8f acc[4];
#pragma unroll
  for (int nt = 0; nt < 4; ++nt)
#pragma unroll
    for (int i = 0; i < 8; ++i) acc[nt][i] = 0.0f;

  // ISA 16-bit fragment layout (wave32):
  // A 16x32: lanes 0-15 hold K 0..7 (e0..7) + 16..23 (e8..15); lanes 16-31: K 8..15 + 24..31.
  // B 32x16: same K split as A (lanes 0-15: K 0..7 + 16..23; lanes 16-31: K 8..15 + 24..31); lane&15 = column.
  const int kL    = (lane < 16) ? 0 : 8;
  const int kB    = (lane < 16) ? 0 : 16;
  const int nLane = lane & 15;

#if USE_TDM
  if (tid < 32) {   // wave 0 is the DMA wave
    tdm_load_tile(&A[(size_t)mBase * K], (unsigned)(size_t)&sA[0][0],
                  (unsigned)K, (unsigned)M, TK, TM, (unsigned)K);
    tdm_load_tile(&Bt[(size_t)nBase * K], (unsigned)(size_t)&sB[0][0],
                  (unsigned)K, (unsigned)N, TK, TN, (unsigned)K);
  }
#endif

  int it = 0;
  for (int k0 = 0; k0 < K; k0 += TK, ++it) {
    const int buf = it & 1;
#if USE_TDM
    if (tid < 32) {
      if (k0 + TK < K) {
        tdm_load_tile(&A[(size_t)mBase * K + k0 + TK], (unsigned)(size_t)&sA[buf ^ 1][0],
                      (unsigned)K, (unsigned)M, TK, TM, (unsigned)K);
        tdm_load_tile(&Bt[(size_t)nBase * K + k0 + TK], (unsigned)(size_t)&sB[buf ^ 1][0],
                      (unsigned)K, (unsigned)N, TK, TN, (unsigned)K);
        __builtin_amdgcn_s_wait_tensorcnt(2);   // current tile's 2 DMAs complete
      } else {
        __builtin_amdgcn_s_wait_tensorcnt(0);
      }
    }
    __syncthreads();
#else
    // synchronous staging fallback (both tiles contiguous: Bt pre-transposed)
#pragma unroll
    for (int i2 = 0; i2 < 2; ++i2) {
      int idx = tid + i2 * 256;        // 0..511
      int r   = idx >> 2;              // 0..127
      int c8  = (idx & 3) * 8;
      *reinterpret_cast<uint4*>(&sA[buf][r * LDA_P + c8]) =
          *reinterpret_cast<const uint4*>(&A[(size_t)(mBase + r) * K + k0 + c8]);
    }
    {
      int r  = tid >> 2;               // 0..63
      int c8 = (tid & 3) * 8;
      *reinterpret_cast<uint4*>(&sB[buf][r * LDB_P + c8]) =
          *reinterpret_cast<const uint4*>(&Bt[(size_t)(nBase + r) * K + k0 + c8]);
    }
    __syncthreads();
#endif

    // A fragment: two contiguous 16B LDS reads
    v8bf alo = *reinterpret_cast<const v8bf*>(&sA[buf][(wave * 16 + nLane) * LDA_P + kL]);
    v8bf ahi = *reinterpret_cast<const v8bf*>(&sA[buf][(wave * 16 + nLane) * LDA_P + kL + 16]);
    v16bf afrag;
#pragma unroll
    for (int i = 0; i < 8; ++i) { afrag[i] = alo[i]; afrag[i + 8] = ahi[i]; }

#pragma unroll
    for (int nt = 0; nt < 4; ++nt) {
      int col = nt * 16 + nLane;
      v8bf blo = *reinterpret_cast<const v8bf*>(&sB[buf][col * LDB_P + kL]);
      v8bf bhi = *reinterpret_cast<const v8bf*>(&sB[buf][col * LDB_P + kL + 16]);
      v16bf bfrag;
#pragma unroll
      for (int i = 0; i < 8; ++i) { bfrag[i] = blo[i]; bfrag[i + 8] = bhi[i]; }
      acc[nt] = __builtin_amdgcn_wmma_f32_16x16x32_bf16(
          false, afrag, false, bfrag, (short)0, acc[nt], false, false);
    }
    __syncthreads();   // all waves done with buf before it is re-filled
  }

  // C/D layout: VGPR g -> M = g (lanes 0-15) or g+8 (lanes 16-31); N = lane&15
#pragma unroll
  for (int nt = 0; nt < 4; ++nt) {
    int colG = nBase + nt * 16 + nLane;
    float bval = bias[colG];
#pragma unroll
    for (int vg = 0; vg < 8; ++vg) {
      int row = mBase + wave * 16 + ((lane < 16) ? vg : vg + 8);
      C[(size_t)row * N + colG] = (bf16_t)(acc[nt][vg] + bval);
    }
  }
}

// ---------------------------------------------------------------- attention @ s=0 (exact entmax-1.5)
__global__ __launch_bounds__(256) void attn_entmax_kernel(
    const bf16_t* __restrict__ xbf, const float* __restrict__ wq, const float* __restrict__ bq,
    const bf16_t* __restrict__ kbf, const bf16_t* __restrict__ vbf,
    const float* __restrict__ temperature, const float* __restrict__ threshold,
    float* __restrict__ ctx) {
  const int bh  = blockIdx.x;
  const int b   = bh >> 4;
  const int h   = bh & 15;
  const int tid = threadIdx.x;

  __shared__ float xrow[HID_D];
  __shared__ float qp[4 * HEAD_D];
  __shared__ float qv[HEAD_D];
  __shared__ float xe[S_LEN];
  __shared__ float xs[S_LEN];
  __shared__ float cum1[S_LEN];
  __shared__ float cum2[S_LEN];
  __shared__ float red1[256];
  __shared__ float red2[256];
  __shared__ int   redi[256];
  __shared__ float ctxp[4 * HEAD_D];
  __shared__ float tstar_s;

  for (int j = tid; j < HID_D; j += 256)
    xrow[j] = (float)xbf[(size_t)(b * S_LEN) * HID_D + j];
  __syncthreads();

  {
    int d = tid & 63, c = tid >> 6;
    float acc = 0.0f;
    int j0 = c * 256;
    for (int j = 0; j < 256; ++j)
      acc += xrow[j0 + j] * wq[(size_t)(j0 + j) * HID_D + h * HEAD_D + d];
    qp[c * HEAD_D + d] = acc;
  }
  __syncthreads();
  if (tid < HEAD_D)
    qv[tid] = bq[h * HEAD_D + tid] + qp[tid] + qp[64 + tid] + qp[128 + tid] + qp[192 + tid];
  __syncthreads();

  const float thr     = threshold[0];
  const float invTemp = 1.0f / temperature[0];

  for (int t = tid; t < S_LEN; t += 256) {
    const bf16_t* kp = &kbf[(size_t)(b * S_LEN + t) * HID_D + h * HEAD_D];
    float s = 0.0f;
    for (int d = 0; d < HEAD_D; ++d) s += (float)kp[d] * qv[d];
    s *= 0.125f;
    s = (s > thr) ? s : 0.0f;
    s *= invTemp;
    xe[t] = 0.5f * s;
  }
  __syncthreads();

  {
    float m = -1e30f;
    for (int t = tid; t < S_LEN; t += 256) m = fmaxf(m, xe[t]);
    red1[tid] = m;
    __syncthreads();
    for (int off = 128; off > 0; off >>= 1) {
      if (tid < off) red1[tid] = fmaxf(red1[tid], red1[tid + off]);
      __syncthreads();
    }
    float mx = red1[0];
    __syncthreads();
    for (int t = tid; t < S_LEN; t += 256) { xe[t] -= mx; xs[t] = xe[t]; }
  }
  __syncthreads();

  for (int k = 2; k <= S_LEN; k <<= 1) {
    for (int j = k >> 1; j > 0; j >>= 1) {
      for (int t = tid; t < S_LEN; t += 256) {
        int ixj = t ^ j;
        if (ixj > t) {
          float a = xs[t], c = xs[ixj];
          bool descBlock = ((t & k) == 0);
          if (descBlock ? (a < c) : (a > c)) { xs[t] = c; xs[ixj] = a; }
        }
      }
      __syncthreads();
    }
  }

  {
    int base = tid * 8;
    float s1 = 0.0f, s2 = 0.0f;
    for (int i = 0; i < 8; ++i) {
      float v = xs[base + i];
      s1 += v; s2 += v * v;
      cum1[base + i] = s1; cum2[base + i] = s2;
    }
    red1[tid] = s1; red2[tid] = s2;
    __syncthreads();
    for (int off = 1; off < 256; off <<= 1) {
      float a1 = (tid >= off) ? red1[tid - off] : 0.0f;
      float a2 = (tid >= off) ? red2[tid - off] : 0.0f;
      __syncthreads();
      red1[tid] += a1; red2[tid] += a2;
      __syncthreads();
    }
    float o1 = (tid > 0) ? red1[tid - 1] : 0.0f;
    float o2 = (tid > 0) ? red2[tid - 1] : 0.0f;
    for (int i = 0; i < 8; ++i) { cum1[base + i] += o1; cum2[base + i] += o2; }
  }
  __syncthreads();

  {
    int cnt = 0;
    for (int t = tid; t < S_LEN; t += 256) {
      float r = (float)(t + 1);
      float mean  = cum1[t] / r;
      float ss    = r * (cum2[t] / r - mean * mean);
      float delta = fmaxf((1.0f - ss) / r, 0.0f);
      float ta    = mean - sqrtf(delta);
      if (ta <= xs[t]) cnt++;
    }
    redi[tid] = cnt;
    __syncthreads();
    for (int off = 128; off > 0; off >>= 1) {
      if (tid < off) redi[tid] += redi[tid + off];
      __syncthreads();
    }
    if (tid == 0) {
      int sup = redi[0]; if (sup < 1) sup = 1;
      int i = sup - 1;
      float r = (float)(i + 1);
      float mean  = cum1[i] / r;
      float ss    = r * (cum2[i] / r - mean * mean);
      float delta = fmaxf((1.0f - ss) / r, 0.0f);
      tstar_s = mean - sqrtf(delta);
    }
  }
  __syncthreads();
  float tstar = tstar_s;

  {
    int d = tid & 63, c = tid >> 6;
    float acc = 0.0f;
    for (int t = c * 512; t < (c + 1) * 512; ++t) {
      float p = xe[t] - tstar;
      if (p > 0.0f)
        acc += p * p * (float)vbf[(size_t)(b * S_LEN + t) * HID_D + h * HEAD_D + d];
    }
    ctxp[c * HEAD_D + d] = acc;
  }
  __syncthreads();
  if (tid < HEAD_D)
    ctx[(size_t)b * HID_D + h * HEAD_D + tid] =
        ctxp[tid] + ctxp[64 + tid] + ctxp[128 + tid] + ctxp[192 + tid];
}

// ---------------------------------------------------------------- o-proj + MLP + sigmoid (2 rows)
__global__ __launch_bounds__(256) void head_mlp_kernel(
    const float* __restrict__ ctx, const float* __restrict__ wo, const float* __restrict__ bo,
    const float* __restrict__ w1, const float* __restrict__ b1,
    const float* __restrict__ w2, const float* __restrict__ b2,
    float* __restrict__ out) {
  const int b   = blockIdx.x;
  const int tid = threadIdx.x;
  __shared__ float ctxr[HID_D];
  __shared__ float orow[HID_D];
  __shared__ float hrow[HID_D / 2];
  for (int j = tid; j < HID_D; j += 256) ctxr[j] = ctx[(size_t)b * HID_D + j];
  __syncthreads();
  for (int n = tid; n < HID_D; n += 256) {
    float acc = bo[n];
    for (int j = 0; j < HID_D; ++j) acc += ctxr[j] * wo[(size_t)j * HID_D + n];
    orow[n] = acc;
  }
  __syncthreads();
  for (int n = tid; n < HID_D / 2; n += 256) {
    float acc = b1[n];
    for (int j = 0; j < HID_D; ++j) acc += orow[j] * w1[(size_t)j * (HID_D / 2) + n];
    hrow[n] = fmaxf(acc, 0.0f);
  }
  __syncthreads();
  for (int n = tid; n < OUT_D; n += 256) {
    float acc = b2[n];
    for (int j = 0; j < HID_D / 2; ++j) acc += hrow[j] * w2[(size_t)j * OUT_D + n];
    out[(size_t)b * OUT_D + n] = 1.0f / (1.0f + expf(-acc));
  }
}

// ---------------------------------------------------------------- launch
extern "C" void kernel_launch(void* const* d_in, const int* in_sizes, int n_in,
                              void* d_out, int out_size, void* d_ws, size_t ws_size,
                              hipStream_t stream) {
  const float* state = (const float*)d_in[0];
  const float* w_emb = (const float*)d_in[1];
  const float* b_emb = (const float*)d_in[2];
  const float* wq    = (const float*)d_in[3];
  const float* bq    = (const float*)d_in[4];
  const float* wk    = (const float*)d_in[5];
  const float* bk    = (const float*)d_in[6];
  const float* wv    = (const float*)d_in[7];
  const float* bv    = (const float*)d_in[8];
  const float* wo    = (const float*)d_in[9];
  const float* bo    = (const float*)d_in[10];
  const float* temp  = (const float*)d_in[11];
  const float* thr   = (const float*)d_in[12];
  const float* w1    = (const float*)d_in[13];
  const float* b1    = (const float*)d_in[14];
  const float* w2    = (const float*)d_in[15];
  const float* b2    = (const float*)d_in[16];
  float* out = (float*)d_out;

  char* ws = (char*)d_ws;
  size_t off = 0;
  auto alloc = [&](size_t bytes) {
    char* p = ws + off;
    off = (off + bytes + 255) & ~(size_t)255;
    return p;
  };
  const int M = BATCH * S_LEN;  // 4096
  bf16_t* state_bf = (bf16_t*)alloc((size_t)M * INPUT_D * 2);
  bf16_t* wembT_bf = (bf16_t*)alloc((size_t)INPUT_D * HID_D * 2);   // [HID][INPUT]
  bf16_t* wkT_bf   = (bf16_t*)alloc((size_t)HID_D * HID_D * 2);     // [HID][HID]
  bf16_t* wvT_bf   = (bf16_t*)alloc((size_t)HID_D * HID_D * 2);     // [HID][HID]
  bf16_t* x_bf     = (bf16_t*)alloc((size_t)M * HID_D * 2);
  bf16_t* k_bf     = (bf16_t*)alloc((size_t)M * HID_D * 2);
  bf16_t* v_bf     = (bf16_t*)alloc((size_t)M * HID_D * 2);
  float*  ctxbuf   = (float*)alloc((size_t)BATCH * HID_D * 4);

  cvt_f32_bf16<<<dim3(512), dim3(256), 0, stream>>>(state, state_bf, M * INPUT_D);
  cvt_f32_bf16_t<<<dim3(256), dim3(256), 0, stream>>>(w_emb, wembT_bf, INPUT_D, HID_D);
  cvt_f32_bf16_t<<<dim3(512), dim3(256), 0, stream>>>(wk, wkT_bf, HID_D, HID_D);
  cvt_f32_bf16_t<<<dim3(512), dim3(256), 0, stream>>>(wv, wvT_bf, HID_D, HID_D);

  // x = state @ w_emb + b_emb   (4096 x 512 x 1024)
  gemm_bf16_wmma<<<dim3(HID_D / TN, M / TM), 256, 0, stream>>>(
      state_bf, wembT_bf, b_emb, x_bf, M, HID_D, INPUT_D);
  // k, v projections (4096 x 1024 x 1024 each)
  gemm_bf16_wmma<<<dim3(HID_D / TN, M / TM), 256, 0, stream>>>(
      x_bf, wkT_bf, bk, k_bf, M, HID_D, HID_D);
  gemm_bf16_wmma<<<dim3(HID_D / TN, M / TM), 256, 0, stream>>>(
      x_bf, wvT_bf, bv, v_bf, M, HID_D, HID_D);

  // attention only for the surviving query position s=0 (32 rows)
  attn_entmax_kernel<<<dim3(BATCH * HEADS_N), 256, 0, stream>>>(
      x_bf, wq, bq, k_bf, v_bf, temp, thr, ctxbuf);

  // o-projection + MLP head
  head_mlp_kernel<<<dim3(BATCH), 256, 0, stream>>>(
      ctxbuf, wo, bo, w1, b1, w2, b2, out);

  (void)in_sizes; (void)n_in; (void)out_size; (void)ws_size;
}
